// TemporalSA_83124797047442
// MI455X (gfx1250) — matrix operand order repaired and checked by reading, not compiled, not run on hardware
//
#include <hip/hip_runtime.h>
#include <stdint.h>

#define B_    8
#define C_    256
#define T_    2048
#define SC_   64
#define DEPTH_ 256
#define OC_   384   // 64(k) + 64(q) + 256(v)

typedef __attribute__((ext_vector_type(16))) __bf16          v16bf;
typedef __attribute__((ext_vector_type(8)))  float           v8f;
typedef __attribute__((ext_vector_type(8)))  unsigned short  u16x8;

union BFrag { v16bf v; u16x8 h[2]; unsigned short u[16]; };

#define WMMA_BF16(A, Bf, Cv) \
  __builtin_amdgcn_wmma_f32_16x16x32_bf16(false, (A), false, (Bf), (short)0, (Cv), false, false)

__device__ __forceinline__ unsigned short f2bf(float f) {
  unsigned int u = __float_as_uint(f);
  u += 0x7FFFu + ((u >> 16) & 1u);   // round-to-nearest-even
  return (unsigned short)(u >> 16);
}

// ---------------------------------------------------------------------------
// Kernel 1: fold BN into conv weights/bias, emit combined W' (bf16) and b' (f32)
// rows [0,64)=k, [64,128)=q, [128,384)=v
// ---------------------------------------------------------------------------
__global__ void k_fold(const float* __restrict__ kW, const float* __restrict__ kb,
                       const float* __restrict__ kg, const float* __restrict__ kbe,
                       const float* __restrict__ kmu, const float* __restrict__ kva,
                       const float* __restrict__ qW, const float* __restrict__ qb,
                       const float* __restrict__ qg, const float* __restrict__ qbe,
                       const float* __restrict__ qmu, const float* __restrict__ qva,
                       const float* __restrict__ vW, const float* __restrict__ vb,
                       const float* __restrict__ vg, const float* __restrict__ vbe,
                       const float* __restrict__ vmu, const float* __restrict__ vva,
                       unsigned short* __restrict__ Wp, float* __restrict__ bp) {
  int oc = blockIdx.x;        // 0..383
  int c  = threadIdx.x;       // 0..255
  const float *W, *bb, *g, *be, *mu, *va; int r;
  if (oc < 64)       { W = kW; bb = kb; g = kg; be = kbe; mu = kmu; va = kva; r = oc; }
  else if (oc < 128) { W = qW; bb = qb; g = qg; be = qbe; mu = qmu; va = qva; r = oc - 64; }
  else               { W = vW; bb = vb; g = vg; be = vbe; mu = vmu; va = vva; r = oc - 128; }
  float inv = g[r] * rsqrtf(va[r] + 1e-5f);
  Wp[oc * C_ + c] = f2bf(W[r * C_ + c] * inv);
  if (c == 0) bp[oc] = (bb[r] - mu[r]) * inv + be[r];
}

// ---------------------------------------------------------------------------
// Kernel 2: x [B,C,T] f32 -> xT [B,T,C] bf16 (LDS tiled transpose, coalesced both sides)
// ---------------------------------------------------------------------------
__global__ void k_xT(const float* __restrict__ x, unsigned short* __restrict__ xT) {
  __shared__ float tile[16][17];
  int b = blockIdx.z, c0 = blockIdx.y * 16, t0 = blockIdx.x * 16;
  int tx = threadIdx.x, ty = threadIdx.y;
  tile[ty][tx] = x[((size_t)b * C_ + (c0 + ty)) * T_ + t0 + tx];
  __syncthreads();
  xT[((size_t)b * T_ + (t0 + ty)) * C_ + c0 + tx] = f2bf(tile[tx][ty]);
}

// ---------------------------------------------------------------------------
// Kernel 3: fused QKV GEMM + bias + ReLU via WMMA bf16.
// Register-blocked 4(t) x 2(oc) tiles per wave: 8 accum tiles (64 VGPRs) and
// 1.5 b128 loads per WMMA -- fits the 256-VGPR direct-address window (no spills).
// mt pairs never straddle type boundaries: pairs 0-1 -> K, 2-3 -> Q, 4-11 -> V.
//   Q,K written as [B,T,64] bf16 ; V written as [B,C,T] bf16 (packed stores).
// ---------------------------------------------------------------------------
__global__ void k_qkv(const unsigned short* __restrict__ xT,
                      const unsigned short* __restrict__ Wp,
                      const float* __restrict__ bp,
                      unsigned short* __restrict__ Qb,
                      unsigned short* __restrict__ Kb,
                      unsigned short* __restrict__ Vb) {
  int w    = blockIdx.x * 4 + (threadIdx.x >> 5);   // global wave id
  int lane = threadIdx.x & 31;
  int ntg = w & 31;              // 32 groups of 4 t-tiles
  int mtg = (w >> 5) % 12;       // 12 groups of 2 oc-tiles
  int b   = w / (32 * 12);       // batch (8)
  int lh = lane & 15, sel = lane >> 4;

  const unsigned short* arow[4];
  const unsigned short* brow[2];
#pragma unroll
  for (int i = 0; i < 4; ++i) {
    int t = (ntg * 4 + i) * 16 + lh;
    arow[i] = xT + ((size_t)b * T_ + t) * C_;
  }
#pragma unroll
  for (int j = 0; j < 2; ++j) {
    int oc = (mtg * 2 + j) * 16 + lh;
    brow[j] = Wp + (size_t)oc * C_;
  }

  v8f acc[2][4] = {};            // [j: oc tile][i: t tile]
  for (int kb = 0; kb < 8; ++kb) {
    int c0 = kb * 32;
    BFrag fa[4], fb[2];
#pragma unroll
    for (int i = 0; i < 4; ++i) {
      fa[i].h[0] = *(const u16x8*)(arow[i] + c0 + sel * 8);        // K {0..7}/{8..15}
      fa[i].h[1] = *(const u16x8*)(arow[i] + c0 + sel * 8 + 16);   // K {16..23}/{24..31}
    }
#pragma unroll
    for (int j = 0; j < 2; ++j) {
      fb[j].h[0] = *(const u16x8*)(brow[j] + c0 + sel * 8);        // K {0..7}/{8..15}
      fb[j].h[1] = *(const u16x8*)(brow[j] + c0 + sel * 8 + 16);
    }
#pragma unroll
    for (int j = 0; j < 2; ++j)
#pragma unroll
      for (int i = 0; i < 4; ++i)
        acc[j][i] = WMMA_BF16(fa[i].v, fb[j].v, acc[j][i]);
  }

#pragma unroll
  for (int j = 0; j < 2; ++j) {
    int mt = mtg * 2 + j;
    int oc = mt * 16 + lh;
    float bias = bp[oc];
#pragma unroll
    for (int i = 0; i < 4; ++i) {
      int nt = ntg * 4 + i;
      if (mt >= 8) {                              // V rows -> [B,C,T], contiguous in t
        int cl = oc - 128;
        u16x8 o;
        for (int r = 0; r < 8; ++r) o[r] = f2bf(fmaxf(acc[j][i][r] + bias, 0.0f));
        *(u16x8*)(Vb + ((size_t)b * C_ + cl) * T_ + nt * 16 + sel * 8) = o;
      } else {                                    // Q/K rows -> [B,T,64]
        unsigned short* dst = (mt < 4) ? Kb : Qb;
        int cl = (mt < 4) ? oc : (oc - 64);
        for (int r = 0; r < 8; ++r) {
          int tt = nt * 16 + sel * 8 + r;
          dst[((size_t)b * T_ + tt) * SC_ + cl] = f2bf(fmaxf(acc[j][i][r] + bias, 0.0f));
        }
      }
    }
  }
}

// ---------------------------------------------------------------------------
// Kernel 4: batch-0 softmax denominators (the reference broadcasts batch 0's
// row sums to every batch).  One wave per 16-query tile of batch 0.
// ---------------------------------------------------------------------------
__global__ void k_denom(const unsigned short* __restrict__ Qb,
                        const unsigned short* __restrict__ Kb,
                        float* __restrict__ denom) {
  __shared__ __align__(16) float sbuf[16][288];
  int qt   = blockIdx.x;
  int lane = threadIdx.x & 31;
  int lh = lane & 15, sel = lane >> 4;
  int tq0   = qt * 16;
  int kt0   = (qt >= 16) ? (qt - 16) : 0;
  int kbase = kt0 * 16;
  int ncols = tq0 + 16 - kbase;        // exact multiple of 16, <= 272
  int nkt   = ncols >> 4;

  const unsigned short* qrow = Qb + ((size_t)tq0 + lh) * SC_;
  BFrag qf0, qf1;
  qf0.h[0] = *(const u16x8*)(qrow + sel * 8);
  qf0.h[1] = *(const u16x8*)(qrow + sel * 8 + 16);
  qf1.h[0] = *(const u16x8*)(qrow + 32 + sel * 8);
  qf1.h[1] = *(const u16x8*)(qrow + 32 + sel * 8 + 16);

  for (int kt = 0; kt < nkt; ++kt) {
    int kk = kbase + kt * 16 + lh;
    const unsigned short* krow = Kb + ((size_t)kk) * SC_;
    BFrag kf0, kf1;
    kf0.h[0] = *(const u16x8*)(krow + sel * 8);
    kf0.h[1] = *(const u16x8*)(krow + sel * 8 + 16);
    kf1.h[0] = *(const u16x8*)(krow + 32 + sel * 8);
    kf1.h[1] = *(const u16x8*)(krow + 32 + sel * 8 + 16);
    v8f s = {};
    s = WMMA_BF16(qf0.v, kf0.v, s);
    s = WMMA_BF16(qf1.v, kf1.v, s);
    for (int r = 0; r < 8; ++r) sbuf[sel * 8 + r][kt * 16 + lh] = s[r] * 0.125f;
  }
  __syncthreads();

  int row = lh, t = tq0 + row;
  int j0 = sel * 144, j1 = j0 + 144;
  if (j1 > ncols) j1 = ncols;
  if (j0 > j1)    j0 = j1;
  float mx = 0.0f;                               // masked zeros participate in max
  for (int j = j0; j < j1; ++j) {
    int k = kbase + j;
    if (k <= t && (t - k) < DEPTH_) mx = fmaxf(mx, sbuf[row][j]);
  }
  mx = fmaxf(mx, __shfl_xor(mx, 16, 32));
  float sum = 0.0f;
  for (int j = j0; j < j1; ++j) {
    int k = kbase + j;
    if (k <= t && (t - k) < DEPTH_) sum += __expf(sbuf[row][j] - mx);
  }
  sum += __shfl_xor(sum, 16, 32);
  if (sel == 0) denom[t] = sum;
}

// ---------------------------------------------------------------------------
// Kernel 5: banded attention + residual.  One wave per (b, 16-query tile).
//   scores (WMMA) -> LDS -> masked softmax with batch-0 denom -> P (bf16, LDS,
//   full 288 cols zero-padded) -> 9 P frags in registers (constant indexing,
//   fully unrolled) -> P x V (WMMA) -> out = x + attn.
// ---------------------------------------------------------------------------
__global__ void k_attn(const unsigned short* __restrict__ Qb,
                       const unsigned short* __restrict__ Kb,
                       const unsigned short* __restrict__ Vb,
                       const float* __restrict__ denom,
                       const float* __restrict__ x,
                       float* __restrict__ out) {
  __shared__ __align__(16) float          sbuf[16][288];
  __shared__ __align__(16) unsigned short pbuf[16][288];
  int qt   = blockIdx.x & 127;
  int b    = blockIdx.x >> 7;
  int lane = threadIdx.x & 31;
  int lh = lane & 15, sel = lane >> 4;
  int tq0   = qt * 16;
  int kt0   = (qt >= 16) ? (qt - 16) : 0;
  int kbase = kt0 * 16;
  int ncols = tq0 + 16 - kbase;        // multiple of 16, <= 272
  int nkt   = ncols >> 4;

  // ---- scores ----
  const unsigned short* qrow = Qb + ((size_t)b * T_ + tq0 + lh) * SC_;
  BFrag qf0, qf1;
  qf0.h[0] = *(const u16x8*)(qrow + sel * 8);
  qf0.h[1] = *(const u16x8*)(qrow + sel * 8 + 16);
  qf1.h[0] = *(const u16x8*)(qrow + 32 + sel * 8);
  qf1.h[1] = *(const u16x8*)(qrow + 32 + sel * 8 + 16);

  for (int kt = 0; kt < nkt; ++kt) {
    int kk = kbase + kt * 16 + lh;
    const unsigned short* krow = Kb + ((size_t)b * T_ + kk) * SC_;
    BFrag kf0, kf1;
    kf0.h[0] = *(const u16x8*)(krow + sel * 8);
    kf0.h[1] = *(const u16x8*)(krow + sel * 8 + 16);
    kf1.h[0] = *(const u16x8*)(krow + 32 + sel * 8);
    kf1.h[1] = *(const u16x8*)(krow + 32 + sel * 8 + 16);
    v8f s = {};
    s = WMMA_BF16(qf0.v, kf0.v, s);
    s = WMMA_BF16(qf1.v, kf1.v, s);
    for (int r = 0; r < 8; ++r) sbuf[sel * 8 + r][kt * 16 + lh] = s[r] * 0.125f;
  }
  __syncthreads();

  // ---- masked softmax (numerator own batch, denominator from batch 0) ----
  // Fill ALL 288 pbuf columns (zeros outside band/ncols) so the P x V loop
  // below can be a constant 9 chunks (keeps pa[] in registers).
  {
    int row = lh, t = tq0 + row;
    int j0 = sel * 144;
    int jm = j0 + 144; if (jm > ncols) jm = ncols;
    float mx = 0.0f;
    for (int j = j0; j < jm; ++j) {
      int k = kbase + j;
      if (k <= t && (t - k) < DEPTH_) mx = fmaxf(mx, sbuf[row][j]);
    }
    mx = fmaxf(mx, __shfl_xor(mx, 16, 32));
    float dinv = 1.0f / (denom[t] + 1e-30f);
    for (int j = j0; j < j0 + 144; ++j) {
      int k = kbase + j;
      float p = 0.0f;
      if (j < ncols && k <= t && (t - k) < DEPTH_)
        p = __expf(sbuf[row][j] - mx) * dinv;
      pbuf[row][j] = f2bf(p);
    }
  }
  __syncthreads();

  // ---- hoist 9 P A-frags into registers (constant indices after unroll) ----
  BFrag pa[9];
#pragma unroll
  for (int kc = 0; kc < 9; ++kc) {
    int jb = kc * 32 + sel * 8;
    pa[kc].h[0] = *(const u16x8*)(&pbuf[lh][jb]);
    pa[kc].h[1] = *(const u16x8*)(&pbuf[lh][jb + 16]);
  }

  // ---- out = P x V (+ residual) ----
  for (int ct = 0; ct < 16; ++ct) {
    int cc = ct * 16 + lh;
    const unsigned short* vbase = Vb + ((size_t)b * C_ + cc) * T_;
    v8f acc = {};
#pragma unroll
    for (int kc = 0; kc < 9; ++kc) {
      BFrag vf;
      int kstart = kbase + kc * 32 + sel * 8;
      if (kstart + 24 > T_) kstart = T_ - 24;     // clamp: shifted V also meets unmasked P
      vf.h[0] = *(const u16x8*)(vbase + kstart);
      vf.h[1] = *(const u16x8*)(vbase + kstart + 16);
      acc = WMMA_BF16(pa[kc].v, vf.v, acc);
    }
    size_t base = ((size_t)b * C_ + cc) * T_ + tq0 + sel * 8;
    float4 x0 = *(const float4*)(x + base);
    float4 x1 = *(const float4*)(x + base + 4);
    float4 o0, o1;
    o0.x = x0.x + acc[0]; o0.y = x0.y + acc[1];
    o0.z = x0.z + acc[2]; o0.w = x0.w + acc[3];
    o1.x = x1.x + acc[4]; o1.y = x1.y + acc[5];
    o1.z = x1.z + acc[6]; o1.w = x1.w + acc[7];
    *(float4*)(out + base)     = o0;
    *(float4*)(out + base + 4) = o1;
  }
}

// ---------------------------------------------------------------------------
// Launch: ws layout (bytes):
//   [0)        Wp   bf16  384*256        = 196608
//   [196608)   bp   f32   384            = 1536
//   [200704)   xT   bf16  8*2048*256     = 8388608
//   [8589312)  Qb   bf16  8*2048*64      = 2097152
//   [10686464) Kb   bf16  8*2048*64      = 2097152
//   [12783616) Vb   bf16  8*256*2048     = 8388608
//   [21172224) denom f32  2048           = 8192     -> total ~20.2 MB
// ---------------------------------------------------------------------------
extern "C" void kernel_launch(void* const* d_in, const int* in_sizes, int n_in,
                              void* d_out, int out_size, void* d_ws, size_t ws_size,
                              hipStream_t stream) {
  (void)in_sizes; (void)n_in; (void)out_size; (void)ws_size;
  const float* x   = (const float*)d_in[0];
  const float* kW  = (const float*)d_in[1];
  const float* kb  = (const float*)d_in[2];
  const float* kg  = (const float*)d_in[3];
  const float* kbe = (const float*)d_in[4];
  const float* kmu = (const float*)d_in[5];
  const float* kva = (const float*)d_in[6];
  const float* qW  = (const float*)d_in[7];
  const float* qb  = (const float*)d_in[8];
  const float* qg  = (const float*)d_in[9];
  const float* qbe = (const float*)d_in[10];
  const float* qmu = (const float*)d_in[11];
  const float* qva = (const float*)d_in[12];
  const float* vW  = (const float*)d_in[13];
  const float* vb  = (const float*)d_in[14];
  const float* vg  = (const float*)d_in[15];
  const float* vbe = (const float*)d_in[16];
  const float* vmu = (const float*)d_in[17];
  const float* vva = (const float*)d_in[18];

  char* ws = (char*)d_ws;
  unsigned short* Wp  = (unsigned short*)(ws);
  float*          bp  = (float*)(ws + 196608);
  unsigned short* xT  = (unsigned short*)(ws + 200704);
  unsigned short* Qb  = (unsigned short*)(ws + 8589312);
  unsigned short* Kb  = (unsigned short*)(ws + 10686464);
  unsigned short* Vb  = (unsigned short*)(ws + 12783616);
  float*          den = (float*)(ws + 21172224);
  float*          out = (float*)d_out;

  k_fold<<<OC_, C_, 0, stream>>>(kW, kb, kg, kbe, kmu, kva,
                                 qW, qb, qg, qbe, qmu, qva,
                                 vW, vb, vg, vbe, vmu, vva, Wp, bp);
  k_xT<<<dim3(T_ / 16, C_ / 16, B_), dim3(16, 16), 0, stream>>>(x, xT);
  k_qkv<<<(B_ * 12 * 32) / 4, 128, 0, stream>>>(xT, Wp, bp, Qb, Kb, Vb);
  k_denom<<<T_ / 16, 32, 0, stream>>>(Qb, Kb, den);
  k_attn<<<B_ * (T_ / 16), 32, 0, stream>>>(Qb, Kb, Vb, den, x, out);
}
